// MultimerTriangleAttention_75282186764818
// MI455X (gfx1250) — matrix operand order repaired and checked by reading, not compiled, not run on hardware
//
#include <hip/hip_runtime.h>
#include <hip/hip_bf16.h>
#include <math.h>

typedef __attribute__((ext_vector_type(16))) __bf16 v16bf;
typedef __attribute__((ext_vector_type(2)))  __bf16 v2bf;
typedef __attribute__((ext_vector_type(8)))  float  v8f;

union Frag16 { unsigned u[8]; v16bf v; };

// f32 -> packed bf16x2 conversion.
// Preference: hw packed cvt > v_perm_b32 byte-select (round-half-up) > manual RNE.
__device__ __forceinline__ unsigned bf16pair(float a, float b) {
#if __has_builtin(__builtin_amdgcn_cvt_pk_bf16_f32)
  v2bf r = __builtin_amdgcn_cvt_pk_bf16_f32(a, b);
  return __builtin_bit_cast(unsigned, r);
#elif __has_builtin(__builtin_amdgcn_perm)
  unsigned ua = __builtin_bit_cast(unsigned, a) + 0x8000u;
  unsigned ub = __builtin_bit_cast(unsigned, b) + 0x8000u;
  // bytes: {ub[31:24], ub[23:16], ua[31:24], ua[23:16]} = {bf16(b), bf16(a)}
  return __builtin_amdgcn_perm(ub, ua, 0x07060302u);
#else
  unsigned ua = __builtin_bit_cast(unsigned, a);
  unsigned ub = __builtin_bit_cast(unsigned, b);
  ua = (ua + 0x7fffu + ((ua >> 16) & 1u)) >> 16;
  ub = ((ub + 0x7fffu + ((ub >> 16) & 1u)) >> 16) << 16;
  return (ua & 0xffffu) | ub;
#endif
}
__device__ __forceinline__ unsigned short bf16s(float a) {
#if __has_builtin(__builtin_amdgcn_cvt_pk_bf16_f32)
  v2bf r = __builtin_amdgcn_cvt_pk_bf16_f32(a, 0.f);
  return (unsigned short)(__builtin_bit_cast(unsigned, r) & 0xffffu);
#else
  unsigned ua = __builtin_bit_cast(unsigned, a);
  return (unsigned short)((ua + 0x8000u) >> 16);
#endif
}

// ---------------------------------------------------------------------------
// GEMM: Y[M,E] = X[M,128] @ W[E,128]^T + bias[E]
// Block tile 128x128, 8 waves each 32x64, K-step 32 (one bf16 WMMA K).
// ---------------------------------------------------------------------------
__global__ __launch_bounds__(256) void gemm_bias_wmma(
    const float* __restrict__ X, const float* __restrict__ W,
    const float* __restrict__ bias, float* __restrict__ Y, int E)
{
  __shared__ unsigned As[128][16];   // [m][k-pair], bf16x2
  __shared__ unsigned Bs[128][16];   // [n][k-pair], bf16x2 (W rows are contiguous in k)
  const int tid  = threadIdx.x;
  const int lane = tid & 31, wid = tid >> 5;
  const int hl   = lane >> 4, l15 = lane & 15;
  const int m0 = blockIdx.x * 128;
  const int n0 = blockIdx.y * 128;
  const int wm = (wid >> 1) * 32;
  const int wn = (wid & 1) * 64;
  const v8f vzero = {0.f,0.f,0.f,0.f,0.f,0.f,0.f,0.f};
  v8f acc[2][4];
  for (int a = 0; a < 2; ++a) for (int b = 0; b < 4; ++b) acc[a][b] = vzero;

  const int srow = tid >> 1;
  const int sc0  = (tid & 1) * 16;

  for (int k0 = 0; k0 < 128; k0 += 32) {
    __syncthreads();
    const float* xs = X + (size_t)(m0 + srow) * 128 + k0 + sc0;
    const float* ws = W + (size_t)(n0 + srow) * 128 + k0 + sc0;
#pragma unroll
    for (int v = 0; v < 4; ++v) {
      float4 f = ((const float4*)xs)[v];
      As[srow][(sc0 >> 1) + 2*v]     = bf16pair(f.x, f.y);
      As[srow][(sc0 >> 1) + 2*v + 1] = bf16pair(f.z, f.w);
      float4 g = ((const float4*)ws)[v];
      Bs[srow][(sc0 >> 1) + 2*v]     = bf16pair(g.x, g.y);
      Bs[srow][(sc0 >> 1) + 2*v + 1] = bf16pair(g.z, g.w);
    }
    __syncthreads();

    Frag16 fa[2], fb[4];
#pragma unroll
    for (int fm = 0; fm < 2; ++fm) {
      int row = wm + fm*16 + l15;
#pragma unroll
      for (int i = 0; i < 8; ++i)
        fa[fm].u[i] = As[row][hl*8 + i];
    }
#pragma unroll
    for (int fn = 0; fn < 4; ++fn) {
      int col = wn + fn*16 + l15;
#pragma unroll
      for (int i = 0; i < 8; ++i)
        fb[fn].u[i] = Bs[col][hl*8 + i];
    }
#pragma unroll
    for (int fm = 0; fm < 2; ++fm)
#pragma unroll
      for (int fn = 0; fn < 4; ++fn)
        acc[fm][fn] = __builtin_amdgcn_wmma_f32_16x16x32_bf16(
            false, fa[fm].v, false, fb[fn].v, (short)0, acc[fm][fn], false, false);
  }

#pragma unroll
  for (int fn = 0; fn < 4; ++fn) {
    int col = n0 + wn + fn*16 + l15;
    float bv = bias[col];
#pragma unroll
    for (int fm = 0; fm < 2; ++fm)
#pragma unroll
      for (int i = 0; i < 8; ++i) {
        int row = m0 + wm + fm*16 + i + 8*hl;
        Y[(size_t)row * E + col] = acc[fm][fn][i] + bv;
      }
  }
}

// ---------------------------------------------------------------------------
// Flash attention over S=256, head_dim=32. One block per (r, h).
// qkv: [R*S, 384] with q at +h*32, k at +128+h*32, v at +256+h*32.
// out: [R*S, 128] head slice at +h*32.
// ---------------------------------------------------------------------------
__global__ __launch_bounds__(256) void attn_wmma(
    const float* __restrict__ qkv, float* __restrict__ out)
{
  __shared__ unsigned Qs[256][16];          // [q][d-pair] bf16x2 (pre-scaled)
  __shared__ unsigned Ks[256][16];          // [key][d-pair]
  __shared__ unsigned short Vt[32][256];    // [d][key]
  __shared__ unsigned short Ps[8][32][32];  // per-wave P chunk [q32][key32]

  const int r = blockIdx.x >> 2, h = blockIdx.x & 3;
  const int tid  = threadIdx.x;
  const int lane = tid & 31, wid = tid >> 5;
  const int hl   = lane >> 4, l15 = lane & 15;
  const float scale = 0.17677669529663687f;   // 1/sqrt(32)

  { // stage q,k,v: thread t handles sequence position t
    const float* row = qkv + (size_t)r * 256 * 384 + (size_t)tid * 384 + h * 32;
#pragma unroll
    for (int c = 0; c < 32; c += 4) {
      float4 f = *(const float4*)(row + c);
      Qs[tid][(c >> 1)]     = bf16pair(f.x * scale, f.y * scale);
      Qs[tid][(c >> 1) + 1] = bf16pair(f.z * scale, f.w * scale);
      float4 g = *(const float4*)(row + 128 + c);
      Ks[tid][(c >> 1)]     = bf16pair(g.x, g.y);
      Ks[tid][(c >> 1) + 1] = bf16pair(g.z, g.w);
      float4 v = *(const float4*)(row + 256 + c);
      Vt[c + 0][tid] = bf16s(v.x);
      Vt[c + 1][tid] = bf16s(v.y);
      Vt[c + 2][tid] = bf16s(v.z);
      Vt[c + 3][tid] = bf16s(v.w);
    }
  }
  __syncthreads();

  // Q fragments for this wave's 32 queries (loaded once)
  Frag16 qa[2];
#pragma unroll
  for (int fm = 0; fm < 2; ++fm) {
    int row = wid * 32 + fm * 16 + l15;
#pragma unroll
    for (int i = 0; i < 8; ++i)
      qa[fm].u[i] = Qs[row][hl*8 + i];
  }

  const v8f vzero = {0.f,0.f,0.f,0.f,0.f,0.f,0.f,0.f};
  v8f acc[2][2];
  float mstate[2][8], lstate[2][8];
#pragma unroll
  for (int fm = 0; fm < 2; ++fm) {
    acc[fm][0] = vzero; acc[fm][1] = vzero;
#pragma unroll
    for (int i = 0; i < 8; ++i) { mstate[fm][i] = -1e30f; lstate[fm][i] = 0.f; }
  }

  for (int kc = 0; kc < 8; ++kc) {
    // K chunk as B fragments (scores = Q @ K^T)
    Frag16 kb[2];
#pragma unroll
    for (int fn = 0; fn < 2; ++fn) {
      int key = kc * 32 + fn * 16 + l15;
#pragma unroll
      for (int i = 0; i < 8; ++i)
        kb[fn].u[i] = Ks[key][hl*8 + i];
    }
    v8f s[2][2];
#pragma unroll
    for (int fm = 0; fm < 2; ++fm)
#pragma unroll
      for (int fn = 0; fn < 2; ++fn)
        s[fm][fn] = __builtin_amdgcn_wmma_f32_16x16x32_bf16(
            false, qa[fm].v, false, kb[fn].v, (short)0, vzero, false, false);

    // online softmax update per row (row = i + 8*hl within frag)
#pragma unroll
    for (int fm = 0; fm < 2; ++fm) {
#pragma unroll
      for (int i = 0; i < 8; ++i) {
        float cm = fmaxf(s[fm][0][i], s[fm][1][i]);
        cm = fmaxf(cm, __shfl_xor(cm, 1));
        cm = fmaxf(cm, __shfl_xor(cm, 2));
        cm = fmaxf(cm, __shfl_xor(cm, 4));
        cm = fmaxf(cm, __shfl_xor(cm, 8));
        float mnew  = fmaxf(mstate[fm][i], cm);
        float alpha = __expf(mstate[fm][i] - mnew);
        float p0 = __expf(s[fm][0][i] - mnew);
        float p1 = __expf(s[fm][1][i] - mnew);
        s[fm][0][i] = p0; s[fm][1][i] = p1;
        float rs = p0 + p1;
        rs += __shfl_xor(rs, 1);
        rs += __shfl_xor(rs, 2);
        rs += __shfl_xor(rs, 4);
        rs += __shfl_xor(rs, 8);
        lstate[fm][i] = lstate[fm][i] * alpha + rs;
        mstate[fm][i] = mnew;
        acc[fm][0][i] *= alpha;
        acc[fm][1][i] *= alpha;
      }
    }

    // stage P (bf16) to LDS so it can be reloaded in A-fragment layout
#pragma unroll
    for (int fm = 0; fm < 2; ++fm)
#pragma unroll
      for (int fn = 0; fn < 2; ++fn)
#pragma unroll
        for (int i = 0; i < 8; ++i)
          Ps[wid][fm*16 + i + 8*hl][fn*16 + l15] = bf16s(s[fm][fn][i]);
    __syncthreads();

    Frag16 pa[2], vb[2];
#pragma unroll
    for (int fm = 0; fm < 2; ++fm) {
      const unsigned short* prow = &Ps[wid][fm*16 + l15][0];
#pragma unroll
      for (int i = 0; i < 8; ++i)
        pa[fm].u[i] = *(const unsigned*)&prow[hl*16 + 2*i];
    }
#pragma unroll
    for (int fn = 0; fn < 2; ++fn) {
      int dim = fn * 16 + l15;
#pragma unroll
      for (int i = 0; i < 8; ++i)
        vb[fn].u[i] = *(const unsigned*)&Vt[dim][kc*32 + hl*16 + 2*i];
    }
#pragma unroll
    for (int fm = 0; fm < 2; ++fm)
#pragma unroll
      for (int fn = 0; fn < 2; ++fn)
        acc[fm][fn] = __builtin_amdgcn_wmma_f32_16x16x32_bf16(
            false, pa[fm].v, false, vb[fn].v, (short)0, acc[fm][fn], false, false);
    __syncthreads();   // protect Ps before next iteration overwrites it
  }

  // normalize and store head output
#pragma unroll
  for (int fm = 0; fm < 2; ++fm)
#pragma unroll
    for (int i = 0; i < 8; ++i) {
      float inv = 1.0f / lstate[fm][i];
      int q = wid * 32 + fm * 16 + i + 8*hl;
      size_t base = ((size_t)r * 256 + q) * 128 + h * 32;
#pragma unroll
      for (int fn = 0; fn < 2; ++fn)
        out[base + fn*16 + l15] = acc[fm][fn][i] * inv;
    }
}

// ---------------------------------------------------------------------------
// LayerNorm: out[rid] = LN(base[rid] + resid[rid or transposed]) * gamma + beta
// ---------------------------------------------------------------------------
__global__ __launch_bounds__(128) void ln_kernel(
    const float* __restrict__ base, const float* __restrict__ resid,
    const float* __restrict__ gamma, const float* __restrict__ beta,
    float* __restrict__ out, int resid_t)
{
  __shared__ float red[128];
  const int rid = blockIdx.x;
  const int d = threadIdx.x;
  size_t ridx;
  if (resid_t) { int i = rid >> 8, j = rid & 255; ridx = ((size_t)j * 256 + i) * 128 + d; }
  else         { ridx = (size_t)rid * 128 + d; }
  float x = base[(size_t)rid * 128 + d] + resid[ridx];

  red[d] = x;
  __syncthreads();
  for (int s = 64; s > 0; s >>= 1) {
    if (d < s) red[d] += red[d + s];
    __syncthreads();
  }
  float mu = red[0] * (1.f / 128.f);
  __syncthreads();
  float c = x - mu;
  red[d] = c * c;
  __syncthreads();
  for (int s = 64; s > 0; s >>= 1) {
    if (d < s) red[d] += red[d + s];
    __syncthreads();
  }
  float var = red[0] * (1.f / 128.f);
  out[(size_t)rid * 128 + d] = c * rsqrtf(var + 1e-5f) * gamma[d] + beta[d];
}

__global__ __launch_bounds__(128) void transpose_kernel(
    const float* __restrict__ in, float* __restrict__ out)
{
  const int rid = blockIdx.x;
  const int i = rid >> 8, j = rid & 255;
  const int d = threadIdx.x;
  out[((size_t)j * 256 + i) * 128 + d] = in[(size_t)rid * 128 + d];
}

__global__ __launch_bounds__(256) void gate_kernel(
    const float* __restrict__ pair, const float* __restrict__ g,
    float* __restrict__ out, int n)
{
  int idx = blockIdx.x * 256 + threadIdx.x;
  if (idx < n) {
    float gv = g[idx];
    out[idx] = pair[idx] * (1.f / (1.f + __expf(-gv)));
  }
}

// ---------------------------------------------------------------------------
extern "C" void kernel_launch(void* const* d_in, const int* in_sizes, int n_in,
                              void* d_out, int out_size, void* d_ws, size_t ws_size,
                              hipStream_t stream) {
  (void)in_sizes; (void)n_in; (void)out_size; (void)ws_size;
  const float* pair_repr = (const float*)d_in[0];
  const float* w_in_s  = (const float*)d_in[2];
  const float* b_in_s  = (const float*)d_in[3];
  const float* w_out_s = (const float*)d_in[4];
  const float* b_out_s = (const float*)d_in[5];
  const float* w_in_e  = (const float*)d_in[6];
  const float* b_in_e  = (const float*)d_in[7];
  const float* w_out_e = (const float*)d_in[8];
  const float* b_out_e = (const float*)d_in[9];
  const float* gamma_s = (const float*)d_in[10];
  const float* beta_s  = (const float*)d_in[11];
  const float* gamma_e = (const float*)d_in[12];
  const float* beta_e  = (const float*)d_in[13];
  const float* w_gate  = (const float*)d_in[14];
  const float* b_gate  = (const float*)d_in[15];
  float* outp = (float*)d_out;

  const size_t NROW = 256u * 256u;            // 65536
  float* ws    = (float*)d_ws;
  float* qkv   = ws;                          // 65536*384
  float* attn  = qkv + NROW * 384;            // 65536*128
  float* pairA = attn + NROW * 128;           // 65536*128
  float* pairB = pairA + NROW * 128;          // 65536*128
  float* proj  = qkv;                         // alias: qkv dead when proj written

  const dim3 gGemmQKV(512, 3), gGemmD(512, 1);
  const dim3 b256(256), b128(128);

  // --- row-wise (starting node) attention ---
  gemm_bias_wmma<<<gGemmQKV, b256, 0, stream>>>(pair_repr, w_in_s, b_in_s, qkv, 384);
  attn_wmma<<<1024, b256, 0, stream>>>(qkv, attn);
  gemm_bias_wmma<<<gGemmD, b256, 0, stream>>>(attn, w_out_s, b_out_s, proj, 128);
  ln_kernel<<<65536, b128, 0, stream>>>(pair_repr, proj, gamma_s, beta_s, pairA, 0);

  // --- column-wise (ending node) attention on transposed pair ---
  transpose_kernel<<<65536, b128, 0, stream>>>(pairA, pairB);
  gemm_bias_wmma<<<gGemmQKV, b256, 0, stream>>>(pairB, w_in_e, b_in_e, qkv, 384);
  attn_wmma<<<1024, b256, 0, stream>>>(qkv, attn);
  gemm_bias_wmma<<<gGemmD, b256, 0, stream>>>(attn, w_out_e, b_out_e, proj, 128);
  ln_kernel<<<65536, b128, 0, stream>>>(pairA, proj, gamma_e, beta_e, pairB, 1);

  // --- gate ---
  gemm_bias_wmma<<<gGemmD, b256, 0, stream>>>(pairB, w_gate, b_gate, qkv, 128);
  gate_kernel<<<(int)((NROW * 128 + 255) / 256), b256, 0, stream>>>(
      pairB, qkv, outp, (int)(NROW * 128));
}
